// ScaledDotProductAttention_33389075759509
// MI455X (gfx1250) — matrix operand order repaired and checked by reading, not compiled, not run on hardware
//
#include <hip/hip_runtime.h>
#include <hip/hip_bf16.h>

// CDNA5 (gfx1250) fused entmax-1.5 attention.
// B=2, H=8, S=2048, D=64, fp32 in/out. bf16 WMMA for both GEMMs,
// fp32 bisection solve for the entmax threshold tau.
// Scale (1/sqrt(D) = 2^-3, exact in bf16) is folded into the Q fragments.

typedef __attribute__((ext_vector_type(16))) __bf16 v16bf;
typedef __attribute__((ext_vector_type(8)))  float  v8f;

#define S_LEN   2048
#define D_DIM   64
#define BM      16
#define SPITCH  2052      // 2048 + 4 pad to stagger LDS banks
#define NWAVES  4
#define NTHREADS (NWAVES * 32)

__device__ __forceinline__ float wave_max(float v) {
#pragma unroll
    for (int m = 16; m >= 1; m >>= 1) v = fmaxf(v, __shfl_xor(v, m, 32));
    return v;
}
__device__ __forceinline__ float wave_sum(float v) {
#pragma unroll
    for (int m = 16; m >= 1; m >>= 1) v += __shfl_xor(v, m, 32);
    return v;
}

__global__ __launch_bounds__(NTHREADS)
void entmax15_attn_kernel(const float* __restrict__ Q,
                          const float* __restrict__ K,
                          const float* __restrict__ V,
                          float* __restrict__ O) {
    extern __shared__ float smem[];
    float* sc   = smem;                        // [BM][SPITCH] scaled scores
    float* red  = smem + BM * SPITCH;          // [NWAVES][BM*D] PV partials
    float* stat = red + NWAVES * BM * D_DIM;   // [BM] row max, [BM] row tau

    const int bh   = blockIdx.y;
    const int q0   = blockIdx.x * BM;
    const int lane = threadIdx.x & 31;
    const int wave = threadIdx.x >> 5;
    const int hl   = lane >> 4;     // which 16-lane half
    const int l16  = lane & 15;

    const float scale = 0.125f;     // 1/sqrt(64), exact power of two

    const float* Qb = Q + (size_t)bh * S_LEN * D_DIM;
    const float* Kb = K + (size_t)bh * S_LEN * D_DIM;
    const float* Vb = V + (size_t)bh * S_LEN * D_DIM;
    float*       Ob = O + (size_t)bh * S_LEN * D_DIM;

    // ---------- Phase A: Q A-fragments (16x32 bf16, two d-halves) ----------
    // A layout: lane l holds row M=l%16; element e maps to
    // K = 16*half + e (contiguous, matches B).
    // Scale folded in here: pre-multiply by 2^-3 (exact) before bf16 rounding.
    v16bf aq[2];
    {
        const float* qrow = Qb + (size_t)(q0 + l16) * D_DIM;
#pragma unroll
        for (int fi = 0; fi < 2; ++fi) {
            const int d0 = fi * 32;
#pragma unroll
            for (int e = 0; e < 16; ++e) {
                int d = d0 + 16 * hl + e;
                aq[fi][e] = (__bf16)(qrow[d] * scale);
            }
        }
    }

    // ---------- Phase 1: scores = (Q*scale) K^T -> LDS ----------
    for (int jt = wave; jt < S_LEN / 16; jt += NWAVES) {
        if (jt + NWAVES < S_LEN / 16)
            __builtin_prefetch(Kb + (size_t)(jt + NWAVES) * 16 * D_DIM, 0, 3);

        const int    key  = jt * 16 + l16;          // B column N = key
        const float* krow = Kb + (size_t)key * D_DIM;
        v8f acc = {0.f, 0.f, 0.f, 0.f, 0.f, 0.f, 0.f, 0.f};
#pragma unroll
        for (int fi = 0; fi < 2; ++fi) {
            // B layout: lanes 0-15 hold K=0..15 (element e=K), lanes 16-31 K=16..31
            const int dbase = fi * 32 + 16 * hl;
            v16bf b;
            const float4* kp = reinterpret_cast<const float4*>(krow + dbase);
#pragma unroll
            for (int c = 0; c < 4; ++c) {
                float4 f = kp[c];
                b[4 * c + 0] = (__bf16)f.x;
                b[4 * c + 1] = (__bf16)f.y;
                b[4 * c + 2] = (__bf16)f.z;
                b[4 * c + 3] = (__bf16)f.w;
            }
            acc = __builtin_amdgcn_wmma_f32_16x16x32_bf16(false, aq[fi], false, b,
                                                          (short)0, acc, false, false);
        }
        // C/D layout: lane l -> N=l%16, VGPR r -> M = r + 8*(l>=16)
#pragma unroll
        for (int r = 0; r < 8; ++r) {
            int m = r + 8 * hl;
            sc[m * SPITCH + jt * 16 + l16] = acc[r];
        }
    }
    __syncthreads();

    // ---------- Phase 2: per-row max + bisection for entmax tau ----------
    // x = (z - max)/2; solve sum(max(x - tau, 0)^2) == 1, tau in [-1, 0].
#pragma unroll 1
    for (int i = 0; i < BM / NWAVES; ++i) {
        const int row = wave * (BM / NWAVES) + i;
        float xl[S_LEN / 32];
        float mx = -3.0e38f;
#pragma unroll
        for (int t = 0; t < S_LEN / 32; ++t) {
            float z = sc[row * SPITCH + lane + 32 * t];
            xl[t] = z;
            mx = fmaxf(mx, z);
        }
        mx = wave_max(mx);
#pragma unroll
        for (int t = 0; t < S_LEN / 32; ++t) xl[t] = (xl[t] - mx) * 0.5f;

        float lo = -1.0f, hi = 0.0f;
#pragma unroll 1
        for (int it = 0; it < 26; ++it) {
            float tau = 0.5f * (lo + hi);
            float s = 0.f;
#pragma unroll
            for (int t = 0; t < S_LEN / 32; ++t) {
                float p = fmaxf(xl[t] - tau, 0.f);
                s = fmaf(p, p, s);
            }
            s = wave_sum(s);
            if (s > 1.0f) lo = tau; else hi = tau;
        }
        if (lane == 0) {
            stat[row]      = mx;
            stat[BM + row] = 0.5f * (lo + hi);
        }
    }
    __syncthreads();

    // ---------- Phase 3: O = P V (P recomputed on the fly) ----------
    const float mrow = stat[l16];        // A row M = l%16
    const float trow = stat[BM + l16];

    v8f oacc[4];
#pragma unroll
    for (int n = 0; n < 4; ++n) {
        v8f z = {0.f, 0.f, 0.f, 0.f, 0.f, 0.f, 0.f, 0.f};
        oacc[n] = z;
    }

    for (int kc = wave; kc < S_LEN / 32; kc += NWAVES) {
        const int k0 = kc * 32;
        // A fragment: probabilities p = max((z-m)/2 - tau, 0)^2, K-dim = keys
        v16bf ap;
#pragma unroll
        for (int e = 0; e < 16; ++e) {
            int key = k0 + 16 * hl + e;
            float x = (sc[l16 * SPITCH + key] - mrow) * 0.5f;
            float p = fmaxf(x - trow, 0.f);
            ap[e] = (__bf16)(p * p);
        }
        // B fragments from V: 4 column tiles of 16 (D=64)
#pragma unroll
        for (int n = 0; n < 4; ++n) {
            const int d = n * 16 + l16;   // B column N = output dim
            v16bf bv;
#pragma unroll
            for (int e = 0; e < 16; ++e) {
                int key = k0 + 16 * hl + e;   // B row K = key
                bv[e] = (__bf16)Vb[(size_t)key * D_DIM + d];
            }
            oacc[n] = __builtin_amdgcn_wmma_f32_16x16x32_bf16(false, ap, false, bv,
                                                              (short)0, oacc[n], false, false);
        }
    }
    // spill per-wave partials, cross-wave reduce through LDS
#pragma unroll
    for (int n = 0; n < 4; ++n) {
#pragma unroll
        for (int r = 0; r < 8; ++r) {
            int m = r + 8 * hl;
            int d = n * 16 + l16;
            red[(wave * BM + m) * D_DIM + d] = oacc[n][r];
        }
    }
    __syncthreads();

    // final reduce + store: thread owns 8 consecutive d at one row m
    {
        const int m  = threadIdx.x >> 3;          // 128 threads -> 16 rows
        const int d0 = (threadIdx.x & 7) * 8;     // 8 consecutive floats
        float4 s0 = {0.f, 0.f, 0.f, 0.f};
        float4 s1 = {0.f, 0.f, 0.f, 0.f};
#pragma unroll
        for (int w = 0; w < NWAVES; ++w) {
            const float4* rp =
                reinterpret_cast<const float4*>(&red[(w * BM + m) * D_DIM + d0]);
            float4 a = rp[0], b = rp[1];
            s0.x += a.x; s0.y += a.y; s0.z += a.z; s0.w += a.w;
            s1.x += b.x; s1.y += b.y; s1.z += b.z; s1.w += b.w;
        }
        float4* op = reinterpret_cast<float4*>(&Ob[(size_t)(q0 + m) * D_DIM + d0]);
        op[0] = s0;
        op[1] = s1;
    }
}

extern "C" void kernel_launch(void* const* d_in, const int* in_sizes, int n_in,
                              void* d_out, int out_size, void* d_ws, size_t ws_size,
                              hipStream_t stream) {
    const float* q = (const float*)d_in[0];
    const float* k = (const float*)d_in[1];
    const float* v = (const float*)d_in[2];
    float* o = (float*)d_out;

    const int bh = in_sizes[0] / (S_LEN * D_DIM);   // B*H = 16
    dim3 grid(S_LEN / BM, bh);
    dim3 block(NTHREADS);
    size_t shmem = (size_t)(BM * SPITCH + NWAVES * BM * D_DIM + 2 * BM) * sizeof(float);
    hipLaunchKernelGGL(entmax15_attn_kernel, grid, block, shmem, stream, q, k, v, o);
}
